// FlareMPNNLSTM_61607010894576
// MI455X (gfx1250) — compile-verified, matrix operand order checked by reading, not run on hardware
//
#include <hip/hip_runtime.h>
#include <hip/hip_bf16.h>

typedef __attribute__((ext_vector_type(16))) _Float16 v16h;
typedef __attribute__((ext_vector_type(8)))  _Float16 v8h;
typedef __attribute__((ext_vector_type(4)))  _Float16 v4h;
typedef __attribute__((ext_vector_type(8)))  float    v8f;
typedef long long i64;

#define T_STEPS 8
#define NN      50000
#define EE      800000
#define FIN     128
#define HH      128
#define KDIM    128          // K is 128 for every GEMM in this model

// ---------------------------------------------------------------- utilities
__global__ __launch_bounds__(256) void zero_f32(float* __restrict__ p, int n) {
  int i = blockIdx.x * 256 + threadIdx.x;
  if (i < n) p[i] = 0.0f;
}

__global__ __launch_bounds__(256) void deg_count(const i64* __restrict__ dst,
                                                 float* __restrict__ deg, int e) {
  int i = blockIdx.x * 256 + threadIdx.x;
  if (i < e) atomicAdd(&deg[(int)dst[i]], 1.0f);
}

__global__ __launch_bounds__(256) void make_nis(const float* __restrict__ deg,
                                                float* __restrict__ nis, int n) {
  int i = blockIdx.x * 256 + threadIdx.x;
  if (i < n) nis[i] = rsqrtf(deg[i] + 1.0f);
}

// agg[n,f] = h[n,f] * nis[n]^2   (self-loop contribution)
__global__ __launch_bounds__(256) void selfloop_init(const float* __restrict__ h,
                                                     const float* __restrict__ nis,
                                                     float* __restrict__ agg, int n) {
  int i = blockIdx.x * 256 + threadIdx.x;
  if (i >= n * HH) return;
  int node = i >> 7;
  float s = nis[node];
  agg[i] = h[i] * s * s;
}

// agg[dst] += h[src] * nis[src]*nis[dst]; 32 lanes x float4 per edge (coalesced)
__global__ __launch_bounds__(256) void scatter_edges(const i64* __restrict__ src,
                                                     const i64* __restrict__ dst,
                                                     const float* __restrict__ h,
                                                     const float* __restrict__ nis,
                                                     float* __restrict__ agg, int e) {
  int gid  = blockIdx.x * 256 + threadIdx.x;
  int edge = gid >> 5;
  if (edge >= e) return;
  int lane = gid & 31;
  int s = (int)src[edge];
  int d = (int)dst[edge];
  float coef = nis[s] * nis[d];
  const float4 v = *(const float4*)(h + (size_t)s * HH + lane * 4);
  float* ad = agg + (size_t)d * HH + lane * 4;
  atomicAdd(ad + 0, v.x * coef);
  atomicAdd(ad + 1, v.y * coef);
  atomicAdd(ad + 2, v.z * coef);
  atomicAdd(ad + 3, v.w * coef);
}

__global__ __launch_bounds__(256) void relu_copy(const float* __restrict__ in,
                                                 float* __restrict__ out, int n) {
  int i = blockIdx.x * 256 + threadIdx.x;
  if (i < n) out[i] = fmaxf(in[i], 0.0f);
}

// ---------------------------------------------------------------- WMMA GEMM
// C[N,P] = A[N,128] * B[128,P] (+bias[P]) (+= existing C if accum)
// bT==0 : B row-major [K][P]   (GCN weights)
// bT==1 : B stored as [P][K]   (w_ih / w_hh: we need B = W^T)
//
// LDS layouts chosen so each WMMA operand fragment = two ds_load_b128:
//   sA  : [row][k]  row stride LDA_S halves (mult of 8 -> 16B-aligned runs)
//   sBt : [col][k]  (B transposed) col stride LDB_S halves
// Per the ISA 16-bit layout, a lane's 16 halves are k = [8kh..8kh+7] and
// [16+8kh..16+8kh+7] at fixed row/col -> contiguous 16B runs in these layouts.
#define LDA_S 40
#define LDB_S 40

__global__ __launch_bounds__(256) void gemm_wmma_f16(const float* __restrict__ A,
                                                     const float* __restrict__ B,
                                                     const float* __restrict__ bias,
                                                     float* __restrict__ C,
                                                     int Nrows, int P,
                                                     int bT, int accum) {
  __shared__ _Float16 sA [128 * LDA_S];
  __shared__ _Float16 sBt[128 * LDB_S];

  const int tid  = threadIdx.x;
  const int lane = tid & 31;
  const int wave = tid >> 5;                 // 0..7 (8 wave32 per block)
  const int row0 = blockIdx.x * 128;
  const int col0 = blockIdx.y * 128;
  const int m    = lane & 15;
  const int kh   = lane >> 4;                // K-half select (ISA 7.12.2)

  v8f acc[8];
#pragma unroll
  for (int i = 0; i < 8; ++i)
    acc[i] = (v8f){0.f, 0.f, 0.f, 0.f, 0.f, 0.f, 0.f, 0.f};

#pragma unroll
  for (int k0 = 0; k0 < KDIM; k0 += 32) {
    // ---- stage A tile (128 rows x 32 k) as f16; float4 reads, v4h stores
#pragma unroll
    for (int it = 0; it < 4; ++it) {
      int li  = tid + 256 * it;              // 0..1023
      int r   = li >> 3;
      int kk0 = (li & 7) * 4;
      int row = row0 + r;
      float4 v = {0.f, 0.f, 0.f, 0.f};
      if (row < Nrows) v = *(const float4*)(A + (size_t)row * KDIM + k0 + kk0);
      v4h hv = {(_Float16)v.x, (_Float16)v.y, (_Float16)v.z, (_Float16)v.w};
      *(v4h*)&sA[r * LDA_S + kk0] = hv;
    }
    // ---- stage B tile transposed into [col][k]
#pragma unroll
    for (int it = 0; it < 4; ++it) {
      int li  = tid + 256 * it;
      int pp  = li >> 3;
      int kk0 = (li & 7) * 4;
      v4h hv;
      if (bT) {  // B is [P][K] row-major: contiguous along k -> float4 read
        float4 v = *(const float4*)(B + (size_t)(col0 + pp) * KDIM + k0 + kk0);
        hv = (v4h){(_Float16)v.x, (_Float16)v.y, (_Float16)v.z, (_Float16)v.w};
      } else {   // B is [K][P]: stride-P reads (weights are tiny, L2-resident)
        const float* bp = B + (size_t)(k0 + kk0) * P + col0 + pp;
        hv = (v4h){(_Float16)bp[0], (_Float16)bp[(size_t)P],
                   (_Float16)bp[2 * (size_t)P], (_Float16)bp[3 * (size_t)P]};
      }
      *(v4h*)&sBt[pp * LDB_S + kk0] = hv;
    }
    __syncthreads();

    if (k0 + 32 < KDIM) {  // hint next A tile into cache (global_prefetch_b8)
      int row = row0 + (tid >> 1);
      if (row < Nrows) __builtin_prefetch(&A[(size_t)row * KDIM + k0 + 32], 0, 1);
    }

    // ---- A fragment: two 16B DS loads
    const _Float16* As = &sA[(wave * 16 + m) * LDA_S + 8 * kh];
    v8h a0 = *(const v8h*)(As);
    v8h a1 = *(const v8h*)(As + 16);
    v16h af = __builtin_shufflevector(a0, a1, 0, 1, 2, 3, 4, 5, 6, 7,
                                      8, 9, 10, 11, 12, 13, 14, 15);

    // ---- 8 column tiles; each B fragment = two 16B DS loads
#pragma unroll
    for (int ct = 0; ct < 8; ++ct) {
      const _Float16* Bs = &sBt[(ct * 16 + m) * LDB_S + 8 * kh];
      v8h b0 = *(const v8h*)(Bs);
      v8h b1 = *(const v8h*)(Bs + 16);
      v16h bf = __builtin_shufflevector(b0, b1, 0, 1, 2, 3, 4, 5, 6, 7,
                                        8, 9, 10, 11, 12, 13, 14, 15);
      acc[ct] = __builtin_amdgcn_wmma_f32_16x16x32_f16(
          false, af, false, bf, (short)0, acc[ct], false, false);
    }
    __syncthreads();
  }

  // ---- store D: VGPR r holds M=r (lanes 0-15) / M=r+8 (lanes 16-31), N=lane&15
  const int rowBase = row0 + wave * 16 + kh * 8;
#pragma unroll
  for (int ct = 0; ct < 8; ++ct) {
    int col  = col0 + ct * 16 + m;
    float bv = bias ? bias[col] : 0.0f;
#pragma unroll
    for (int r = 0; r < 8; ++r) {
      int row = rowBase + r;
      if (row < Nrows) {
        size_t idx = (size_t)row * P + col;
        float v = acc[ct][r] + bv;
        if (accum) v += C[idx];
        C[idx] = v;
      }
    }
  }
}

// ---------------------------------------------------------------- LSTM cell
__global__ __launch_bounds__(256) void lstm_pointwise(const float* __restrict__ g,
                                                      float* __restrict__ h,
                                                      float* __restrict__ c, int n) {
  int i = blockIdx.x * 256 + threadIdx.x;
  if (i >= n * HH) return;
  int node = i >> 7, j = i & 127;
  const float* gn = g + (size_t)node * 4 * HH;
  float gi = gn[j];
  float gf = gn[HH + j];
  float gg = gn[2 * HH + j];
  float go = gn[3 * HH + j];
  float si = 1.0f / (1.0f + __expf(-gi));
  float sf = 1.0f / (1.0f + __expf(-gf));
  float so = 1.0f / (1.0f + __expf(-go));
  float cc = sf * c[i] + si * tanhf(gg);
  c[i] = cc;
  h[i] = so * tanhf(cc);
}

// out[n] = dot(h2[n,:], w) + b  — one wave32 per node, shuffle reduce
__global__ __launch_bounds__(256) void out_proj(const float* __restrict__ h2,
                                                const float* __restrict__ w,
                                                const float* __restrict__ b,
                                                float* __restrict__ out, int n) {
  int gid  = blockIdx.x * 256 + threadIdx.x;
  int node = gid >> 5;
  if (node >= n) return;
  int lane = gid & 31;
  const float* hn = h2 + (size_t)node * HH;
  float s = 0.0f;
#pragma unroll
  for (int j = lane; j < HH; j += 32) s += hn[j] * w[j];
#pragma unroll
  for (int off = 16; off > 0; off >>= 1) s += __shfl_down(s, off, 32);
  if (lane == 0) out[node] = s + b[0];
}

// ---------------------------------------------------------------- launcher
extern "C" void kernel_launch(void* const* d_in, const int* in_sizes, int n_in,
                              void* d_out, int out_size, void* d_ws, size_t ws_size,
                              hipStream_t stream) {
  const float* x    = (const float*)d_in[0];
  const i64*   ei   = (const i64*)d_in[1];
  const float* gw1  = (const float*)d_in[2];
  const float* gb1  = (const float*)d_in[3];
  const float* gw2  = (const float*)d_in[4];
  const float* gb2  = (const float*)d_in[5];
  const float* wih1 = (const float*)d_in[6];
  const float* whh1 = (const float*)d_in[7];
  const float* bih1 = (const float*)d_in[8];
  const float* bhh1 = (const float*)d_in[9];
  const float* wih2 = (const float*)d_in[10];
  const float* whh2 = (const float*)d_in[11];
  const float* bih2 = (const float*)d_in[12];
  const float* bhh2 = (const float*)d_in[13];
  const float* outw = (const float*)d_in[14];
  const float* outb = (const float*)d_in[15];

  float* outs = (float*)d_out;
  float* h1 = outs + (size_t)T_STEPS * NN;
  float* c1 = h1 + (size_t)NN * HH;
  float* h2 = c1 + (size_t)NN * HH;
  float* c2 = h2 + (size_t)NN * HH;

  float* ws    = (float*)d_ws;
  float* feats = ws;  ws += (size_t)T_STEPS * NN * HH;
  float* hbuf  = ws;  ws += (size_t)NN * HH;
  float* abuf  = ws;  ws += (size_t)NN * HH;
  float* gbuf  = ws;  ws += (size_t)NN * 4 * HH;
  float* deg   = ws;  ws += NN;
  float* nis   = ws;  ws += NN;

  const int NH = NN * HH;
  // zero initial LSTM states (h1,c1,h2,c2 are contiguous in d_out)
  zero_f32<<<(4 * NH + 255) / 256, 256, 0, stream>>>(h1, 4 * NH);

  dim3 gGcn((NN + 127) / 128, 1);
  dim3 gGate((NN + 127) / 128, 4);  // P = 512

  // ---- GCN per snapshot
  for (int t = 0; t < T_STEPS; ++t) {
    const i64* src = ei + (size_t)t * 2 * EE;
    const i64* dst = src + EE;

    zero_f32<<<(NN + 255) / 256, 256, 0, stream>>>(deg, NN);
    deg_count<<<(EE + 255) / 256, 256, 0, stream>>>(dst, deg, EE);
    make_nis<<<(NN + 255) / 256, 256, 0, stream>>>(deg, nis, NN);

    // layer 1
    gemm_wmma_f16<<<gGcn, 256, 0, stream>>>(x + (size_t)t * NN * FIN, gw1, gb1,
                                            hbuf, NN, HH, 0, 0);
    selfloop_init<<<(NH + 255) / 256, 256, 0, stream>>>(hbuf, nis, abuf, NN);
    scatter_edges<<<(EE * 32 + 255) / 256, 256, 0, stream>>>(src, dst, hbuf, nis, abuf, EE);
    relu_copy<<<(NH + 255) / 256, 256, 0, stream>>>(abuf, hbuf, NH);
    // layer 2
    gemm_wmma_f16<<<gGcn, 256, 0, stream>>>(hbuf, gw2, gb2, abuf, NN, HH, 0, 0);
    selfloop_init<<<(NH + 255) / 256, 256, 0, stream>>>(abuf, nis, hbuf, NN);
    scatter_edges<<<(EE * 32 + 255) / 256, 256, 0, stream>>>(src, dst, abuf, nis, hbuf, EE);
    relu_copy<<<(NH + 255) / 256, 256, 0, stream>>>(hbuf, feats + (size_t)t * NH, NH);
  }

  // ---- 2-layer LSTM scan over time
  for (int t = 0; t < T_STEPS; ++t) {
    const float* ft = feats + (size_t)t * NH;
    // layer 1 gates: g = ft @ wih1^T + bih1 ; g += h1 @ whh1^T + bhh1
    gemm_wmma_f16<<<gGate, 256, 0, stream>>>(ft, wih1, bih1, gbuf, NN, 4 * HH, 1, 0);
    gemm_wmma_f16<<<gGate, 256, 0, stream>>>(h1, whh1, bhh1, gbuf, NN, 4 * HH, 1, 1);
    lstm_pointwise<<<(NH + 255) / 256, 256, 0, stream>>>(gbuf, h1, c1, NN);
    // layer 2
    gemm_wmma_f16<<<gGate, 256, 0, stream>>>(h1, wih2, bih2, gbuf, NN, 4 * HH, 1, 0);
    gemm_wmma_f16<<<gGate, 256, 0, stream>>>(h2, whh2, bhh2, gbuf, NN, 4 * HH, 1, 1);
    lstm_pointwise<<<(NH + 255) / 256, 256, 0, stream>>>(gbuf, h2, c2, NN);
    // head
    out_proj<<<(NN * 32 + 255) / 256, 256, 0, stream>>>(h2, outw, outb,
                                                        outs + (size_t)t * NN, NN);
  }
}
